// DynamicRouting_12713103197276
// MI455X (gfx1250) — matrix operand order repaired and checked by reading, not compiled, not run on hardware
//
#include <hip/hip_runtime.h>

typedef __attribute__((ext_vector_type(16))) __bf16         v16bf;
typedef __attribute__((ext_vector_type(8)))  float          v8f;
typedef __attribute__((ext_vector_type(4)))  float          v4f;
typedef __attribute__((ext_vector_type(16))) unsigned short v16us;
typedef __attribute__((ext_vector_type(8)))  unsigned short v8us;
typedef __attribute__((ext_vector_type(4)))  unsigned short v4us;

static constexpr int CCAPS = 512;
static constexpr int KCAPS = 64;
static constexpr int HDIM  = 1024;
static constexpr int MROWS = CCAPS * KCAPS;   // 32768

// ---------------------------------------------------------------------------
// fp32 -> bf16 (round-to-nearest-even), 4 elements per thread
// ---------------------------------------------------------------------------
__device__ __forceinline__ unsigned short f2bf(float f) {
  unsigned u = __builtin_bit_cast(unsigned, f);
  u += 0x7FFFu + ((u >> 16) & 1u);
  return (unsigned short)(u >> 16);
}

__global__ __launch_bounds__(256)
void cvt4_f32_bf16(const float* __restrict__ in, unsigned short* __restrict__ out, int n4) {
  int i = blockIdx.x * blockDim.x + threadIdx.x;
  if (i < n4) {
    v4f v = *(const v4f*)(in + (size_t)i * 4);
    v4us o;
#pragma unroll
    for (int j = 0; j < 4; ++j) o[j] = f2bf(v[j]);
    *(v4us*)(out + (size_t)i * 4) = o;
  }
}

// ---------------------------------------------------------------------------
// hat[M, H] = Xb[M, H] @ Wb[H, H]^T   (bf16 in, fp32 accumulate via WMMA)
// One wave computes a 32x32 tile: 2x2 fragments, K stepped by 32.
// ---------------------------------------------------------------------------
__device__ __forceinline__ v16bf load_frag(const unsigned short* p, int second_off) {
  v8us lo = *(const v8us*)(p);
  v8us hi = *(const v8us*)(p + second_off);
  v16us t;
#pragma unroll
  for (int i = 0; i < 8; ++i) { t[i] = lo[i]; t[i + 8] = hi[i]; }
  return __builtin_bit_cast(v16bf, t);
}

__global__ __launch_bounds__(256)
void wmma_gemm_bf16(const unsigned short* __restrict__ A,   // [MROWS, H] bf16 bits
                    const unsigned short* __restrict__ B,   // [H, H] bf16 bits (= W, row o, col h)
                    float* __restrict__ D)                  // [MROWS, H] fp32
{
  const int lane  = threadIdx.x & 31;
  const int wave  = blockIdx.x * (blockDim.x >> 5) + (threadIdx.x >> 5);
  const int ntile = HDIM / 32;                 // 32 column tiles
  const int tm    = (wave / ntile) * 32;
  const int tn    = (wave % ntile) * 32;
  const int half  = lane >> 4;                 // lane group: 0 or 1
  const int r     = lane & 15;

  v8f acc[2][2];
#pragma unroll
  for (int mi = 0; mi < 2; ++mi)
#pragma unroll
    for (int ni = 0; ni < 2; ++ni) acc[mi][ni] = (v8f)0.0f;

  for (int k0 = 0; k0 < HDIM; k0 += 32) {
    v16bf af[2], bf[2];
#pragma unroll
    for (int mi = 0; mi < 2; ++mi) {
      // A 16x32 bf16 layout: lanes 0-15 hold row r, K {0..7, 16..23};
      //                      lanes 16-31 hold row r, K {8..15, 24..31}
      const unsigned short* ap = A + (size_t)(tm + mi * 16 + r) * HDIM + k0 + half * 8;
      af[mi] = load_frag(ap, 16);
      __builtin_prefetch(ap + 32, 0, 0);       // next K-step -> global_prefetch
    }
#pragma unroll
    for (int ni = 0; ni < 2; ++ni) {
      // B 32x16 bf16 layout: lane holds column n = r, K {0..7, 16..23} (lanes 0-15) / {8..15, 24..31} (lanes 16-31).
      // Column n of B == row (tn+n) of W, contiguous in k -> two b128 loads.
      const unsigned short* bp = B + (size_t)(tn + ni * 16 + r) * HDIM + k0 + half * 8;
      bf[ni] = load_frag(bp, 16);
      __builtin_prefetch(bp + 32, 0, 0);
    }
#pragma unroll
    for (int mi = 0; mi < 2; ++mi)
#pragma unroll
      for (int ni = 0; ni < 2; ++ni)
        acc[mi][ni] = __builtin_amdgcn_wmma_f32_16x16x32_bf16(
            /*neg_a=*/false, af[mi], /*neg_b=*/false, bf[ni],
            /*c_mod=*/(short)0, acc[mi][ni], /*reuse_a=*/false, /*reuse_b=*/false);
  }

  // C/D layout: VGPR v, lanes 0-15 -> (M=v, N=lane); lanes 16-31 -> (M=v+8, N=lane-16)
#pragma unroll
  for (int mi = 0; mi < 2; ++mi)
#pragma unroll
    for (int ni = 0; ni < 2; ++ni)
#pragma unroll
      for (int v = 0; v < 8; ++v)
        D[(size_t)(tm + mi * 16 + half * 8 + v) * HDIM + tn + ni * 16 + r] = acc[mi][ni][v];
}

// ---------------------------------------------------------------------------
// Per-capsule: d = softmax(b[c,:]); c_hat = sum_k d[k]*hat[c,k,:]; c = squash
// One block (256 threads) per capsule. iter==0 -> b treated as zeros.
// ---------------------------------------------------------------------------
__global__ __launch_bounds__(256)
void routing_reduce_squash(const float* __restrict__ hat,
                           const float* __restrict__ blog,
                           float* __restrict__ cOut, int iter)
{
  const int c   = blockIdx.x;
  const int tid = threadIdx.x;
  __shared__ float s[KCAPS];
  __shared__ float d[KCAPS];
  __shared__ float part[8];
  __shared__ float nshared;

  if (tid < KCAPS) s[tid] = (iter == 0) ? 0.0f : blog[c * KCAPS + tid];
  __syncthreads();

  if (tid < 32) {                         // max over 64
    float m = fmaxf(s[tid], s[tid + 32]);
#pragma unroll
    for (int off = 16; off; off >>= 1) m = fmaxf(m, __shfl_down(m, off));
    if (tid == 0) part[0] = m;
  }
  __syncthreads();
  const float mx = part[0];
  if (tid < KCAPS) d[tid] = __expf(s[tid] - mx);
  __syncthreads();
  if (tid < 32) {                         // sum over 64
    float sm = d[tid] + d[tid + 32];
#pragma unroll
    for (int off = 16; off; off >>= 1) sm += __shfl_down(sm, off);
    if (tid == 0) part[0] = sm;
  }
  __syncthreads();
  const float inv = 1.0f / part[0];
  if (tid < KCAPS) d[tid] *= inv;
  __syncthreads();

  // weighted sum over K: thread tid handles h in [4*tid, 4*tid+4)
  v4f acc = (v4f)0.0f;
  const float* hp = hat + (size_t)c * KCAPS * HDIM + tid * 4;
#pragma unroll 4
  for (int k = 0; k < KCAPS; ++k) {
    v4f v = *(const v4f*)(hp + (size_t)k * HDIM);
    acc += v * d[k];
  }

  // block reduction of ||c_hat||^2
  float p = acc[0] * acc[0] + acc[1] * acc[1] + acc[2] * acc[2] + acc[3] * acc[3];
#pragma unroll
  for (int off = 16; off; off >>= 1) p += __shfl_down(p, off);
  if ((tid & 31) == 0) part[tid >> 5] = p;
  __syncthreads();
  if (tid == 0) {
    float t = 0.0f;
#pragma unroll
    for (int w = 0; w < 8; ++w) t += part[w];
    nshared = t;
  }
  __syncthreads();
  const float nsq   = nshared;
  const float scale = (nsq / (1.0f + nsq)) * rsqrtf(nsq + 1e-9f);
  *(v4f*)(cOut + (size_t)c * HDIM + tid * 4) = acc * scale;
}

// ---------------------------------------------------------------------------
// b[c,k] (+)= dot(hat[c,k,:], c[c,:]) over H=1024. One wave per (c,k).
// ---------------------------------------------------------------------------
__global__ __launch_bounds__(256)
void routing_agreement(const float* __restrict__ hat,
                       const float* __restrict__ cVec,
                       float* __restrict__ blog, int accumulate)
{
  const int wave = blockIdx.x * (blockDim.x >> 5) + (threadIdx.x >> 5); // = c*64 + k
  const int lane = threadIdx.x & 31;
  const int c    = wave >> 6;
  const float* hp = hat + (size_t)wave * HDIM;
  const float* cp = cVec + (size_t)c * HDIM;

  float s = 0.0f;
#pragma unroll
  for (int i = 0; i < 8; ++i) {
    const int h = lane * 4 + i * 128;
    v4f a = *(const v4f*)(hp + h);
    v4f b = *(const v4f*)(cp + h);
    s += a[0] * b[0] + a[1] * b[1] + a[2] * b[2] + a[3] * b[3];
  }
#pragma unroll
  for (int off = 16; off; off >>= 1) s += __shfl_down(s, off);
  if (lane == 0) blog[wave] = accumulate ? (blog[wave] + s) : s;
}

// ---------------------------------------------------------------------------
extern "C" void kernel_launch(void* const* d_in, const int* in_sizes, int n_in,
                              void* d_out, int out_size, void* d_ws, size_t ws_size,
                              hipStream_t stream) {
  const float* X = (const float*)d_in[0];   // [512, 64, 1024] fp32
  const float* W = (const float*)d_in[1];   // [1024, 1024] fp32
  float* out     = (float*)d_out;           // [512, 1024] fp32

  char* ws = (char*)d_ws;
  const size_t hat_bytes = (size_t)MROWS * HDIM * sizeof(float);      // 128 MB
  const size_t xb_bytes  = (size_t)MROWS * HDIM * sizeof(unsigned short); // 64 MB
  const size_t wb_bytes  = (size_t)HDIM * HDIM * sizeof(unsigned short);  // 2 MB
  float*          hat = (float*)ws;
  unsigned short* Xb  = (unsigned short*)(ws + hat_bytes);
  unsigned short* Wb  = (unsigned short*)(ws + hat_bytes + xb_bytes);
  float*          bb  = (float*)(ws + hat_bytes + xb_bytes + wb_bytes);

  // 1) fp32 -> bf16 operand conversion (both bf16 buffers then live in L2)
  {
    const int n4x = (MROWS * HDIM) / 4;   // 8388608
    cvt4_f32_bf16<<<(n4x + 255) / 256, 256, 0, stream>>>(X, Xb, n4x);
    const int n4w = (HDIM * HDIM) / 4;    // 262144
    cvt4_f32_bf16<<<(n4w + 255) / 256, 256, 0, stream>>>(W, Wb, n4w);
  }

  // 2) hat = Xb @ Wb^T via bf16 WMMA, fp32 accumulate
  {
    const int waves  = (MROWS / 32) * (HDIM / 32);  // 32768
    const int blocks = waves / 8;                   // 4096 blocks of 8 waves
    wmma_gemm_bf16<<<blocks, 256, 0, stream>>>(Xb, Wb, hat);
  }

  // 3) routing iterations (hat = 128 MB, resident in 192 MB L2)
  for (int it = 0; it < 3; ++it) {
    routing_reduce_squash<<<CCAPS, 256, 0, stream>>>(hat, bb, out, it);
    if (it < 2) {
      const int blocks = (CCAPS * KCAPS) / 8;       // 4096 blocks, 1 wave per (c,k)
      routing_agreement<<<blocks, 256, 0, stream>>>(hat, out, bb, it > 0 ? 1 : 0);
    }
  }
}
